// KGATPGExp_5050881540693
// MI455X (gfx1250) — matrix operand order repaired and checked by reading, not compiled, not run on hardware
//
#include <hip/hip_runtime.h>
#include <hip/hip_bf16.h>
#include <math.h>

typedef __attribute__((ext_vector_type(16))) _Float16 v16h;
typedef __attribute__((ext_vector_type(8)))  _Float16 h8;
typedef __attribute__((ext_vector_type(4)))  _Float16 h4;
typedef __attribute__((ext_vector_type(8)))  float    v8f;
typedef __attribute__((ext_vector_type(4)))  unsigned int u4v;

#define HID 64
#define FD  128
#define WAVES_PER_BLOCK 8

struct BPair { u4v lo, hi; };   // 32 bytes == v16h

// ---------------- Kernel 0a: embed fp32 -> f16 ----------------
__global__ void cvt_embed_kernel(const float* __restrict__ embed,
                                 _Float16* __restrict__ embH, long long total) {
    long long i = ((long long)blockIdx.x * blockDim.x + threadIdx.x) * 4;
    if (i + 3 < total) {
        float4 v = *(const float4*)(embed + i);
        h4 o; o[0] = (_Float16)v.x; o[1] = (_Float16)v.y;
              o[2] = (_Float16)v.z; o[3] = (_Float16)v.w;
        *(h4*)(embH + i) = o;
    }
}

// ---------------- Kernel 0b: pack W1[0:256,0:64] into WMMA B layout ----------------
// idx = ((kt*4 + nt)*32 + lane)*8 + j  holds f16 pair (K = kt*32 + (lane>>4)*16 + 2j, K+1)
// at N = nt*16 + (lane&15).  (this code's contiguous K order, matched by A: lanes 0-15 hold K 0-15, lanes 16-31 hold K 16-31.)
__global__ void pack_w1_kernel(const float* __restrict__ W1,
                               unsigned int* __restrict__ w1pack) {
    int tid  = blockIdx.x * blockDim.x + threadIdx.x;   // 0..8191
    int j    = tid & 7;
    int lane = (tid >> 3) & 31;
    int nt   = (tid >> 8) & 3;
    int kt   = tid >> 10;
    int n = nt * 16 + (lane & 15);
    int k = kt * 32 + (lane >> 4) * 16 + 2 * j;
    union { _Float16 h[2]; unsigned int u; } p;
    p.h[0] = (_Float16)W1[(k + 0) * HID + n];
    p.h[1] = (_Float16)W1[(k + 1) * HID + n];
    w1pack[tid] = p.u;
}

// ---------------- Kernel 0c: c = b1 + src@W1[2F:3F] + dst@W1[3F:4F] ----------------
__global__ void const_vec_kernel(const float* __restrict__ embed,
                                 const float* __restrict__ W1,
                                 const float* __restrict__ b1,
                                 const int* __restrict__ srcp,
                                 const int* __restrict__ dstp,
                                 float* __restrict__ cvec) {
    int n = threadIdx.x;                 // 64 threads
    const float* se = embed + (long long)srcp[0] * FD;
    const float* de = embed + (long long)dstp[0] * FD;
    float s = b1[n];
    for (int f = 0; f < FD; ++f) {
        s += se[f] * W1[(2 * FD + f) * HID + n];
        s += de[f] * W1[(3 * FD + f) * HID + n];
    }
    cvec[n] = s;
}

// ---------------- Main WMMA edge-MLP kernel ----------------
// The per-lane LDS offset is pinned by an empty *volatile* inline asm inside
// the tile loop: the B addresses become opaque per-iteration values, so the
// compiler cannot hoist the (loop-invariant) LDS B-tiles into 256 VGPRs
// (round 1: occupancy 3) nor spill them to scratch under a register cap
// (round 2), and — unlike a volatile load (round 3) — the accesses remain
// ordinary ds_load_b128 with normal scheduling and staged s_wait_dscnt.
__global__ __launch_bounds__(256) void edge_mlp_kernel(
    const _Float16* __restrict__ embH,
    const unsigned int* __restrict__ w1pack,
    const float* __restrict__ cvec,
    const float* __restrict__ W2,
    const float* __restrict__ b2p,
    const float* __restrict__ noise,
    const int* __restrict__ col,
    const int* __restrict__ row,
    float* __restrict__ out,
    int E, int ntiles)
{
    __shared__ unsigned int ldsB[8192];            // 32 KB: full packed W1
    for (int i = threadIdx.x; i < 8192; i += 256) ldsB[i] = w1pack[i];
    __syncthreads();

    const int lane = threadIdx.x & 31;
    const int wid  = threadIdx.x >> 5;
    const int lh   = lane >> 4;                    // wave half (K-group select)
    const int nlo  = lane & 15;

    // loop-invariant per-lane constants (n = 16t + nlo)
    float cv[4], w2v[4];
#pragma unroll
    for (int t = 0; t < 4; ++t) { cv[t] = cvec[t * 16 + nlo]; w2v[t] = W2[t * 16 + nlo]; }
    const float b2v = b2p[0];

    for (int tile = blockIdx.x * WAVES_PER_BLOCK + wid; tile < ntiles;
         tile += gridDim.x * WAVES_PER_BLOCK) {
        const int ebase = tile * 16;
        int e = ebase + nlo; if (e >= E) e = E - 1;     // clamp: keep EXEC all-ones
        const int cbase = col[e] * FD;
        const int rbase = row[e] * FD;

        // -------- prefetch the whole 16x256 A panel: 16 x b128 = 64 VGPRs --------
        h8 ga[16];
#pragma unroll
        for (int kt = 0; kt < 8; ++kt) {
            const int base = (kt < 4) ? cbase : rbase;
            const int fb   = (kt & 3) * 32 + lh * 16;
            ga[2 * kt]     = *(const h8*)(embH + base + fb);
            ga[2 * kt + 1] = *(const h8*)(embH + base + fb + 8);
        }

        // opaque per-iteration LDS byte offset of this lane's B row
        unsigned bpin = (unsigned)lane * 32u;
        asm volatile("" : "+v"(bpin));              // no code; defeats LICM on B loads
        const char* bbase = (const char*)ldsB + bpin;

        v8f acc[4];
#pragma unroll
        for (int t = 0; t < 4; ++t) acc[t] = (v8f){0.f,0.f,0.f,0.f,0.f,0.f,0.f,0.f};

#pragma unroll
        for (int kt = 0; kt < 8; ++kt) {
            v16h a;
#pragma unroll
            for (int i = 0; i < 8; ++i) { a[i] = ga[2 * kt][i]; a[8 + i] = ga[2 * kt + 1][i]; }
#pragma unroll
            for (int nt = 0; nt < 4; ++nt) {
                const unsigned boff = (unsigned)(kt * 4 + nt) * 1024u;   // ds imm offset
                BPair t;
                t.lo = *(const u4v*)(bbase + boff);         // ds_load_b128
                t.hi = *(const u4v*)(bbase + boff + 16);    // ds_load_b128
                v16h b = __builtin_bit_cast(v16h, t);
                acc[nt] = __builtin_amdgcn_wmma_f32_16x16x32_f16(
                    false, a, false, b, (short)0, acc[nt], false, false);
            }
        }

        // Epilogue: w[m] = b2 + sum_n relu(h_pre[m][n] + c[n]) * W2[n]
        // C layout: VGPR r, lanes 0-15 -> M=r, lanes 16-31 -> M=r+8; N = nlo + 16*nt
        float s[8];
#pragma unroll
        for (int r = 0; r < 8; ++r) {
            float v = 0.f;
#pragma unroll
            for (int t = 0; t < 4; ++t) {
                float h = acc[t][r] + cv[t];
                h = h > 0.f ? h : 0.f;
                v += h * w2v[t];
            }
            v += __shfl_xor(v, 1, 32);
            v += __shfl_xor(v, 2, 32);
            v += __shfl_xor(v, 4, 32);
            v += __shfl_xor(v, 8, 32);   // masks < 16: stays within the 16-lane half
            s[r] = v;
        }
        // every lane of half h now holds s[r] = w[r + 8h]; select r = lane&7
        const int rs = lane & 7;
        float w = s[0];
        w = (rs == 1) ? s[1] : w;
        w = (rs == 2) ? s[2] : w;
        w = (rs == 3) ? s[3] : w;
        w = (rs == 4) ? s[4] : w;
        w = (rs == 5) ? s[5] : w;
        w = (rs == 6) ? s[6] : w;
        w = (rs == 7) ? s[7] : w;
        w += b2v;

        if (nlo < 8) {                       // lanes 0-7 -> edges 0-7, lanes 16-23 -> edges 8-15
            int m  = lh * 8 + rs;
            int eo = ebase + m;
            if (eo < E) {
                float u = noise[eo];
                float x = logf(u) - log1pf(-u) + w;     // logit(noise) + w, BETA=1
                out[eo] = 1.f / (1.f + expf(-x));
            }
        }
    }
}

// ---------------- Fallback (ws too small): scalar fp32 ----------------
__global__ void fallback_kernel(const float* embed, const float* W1, const float* b1,
                                const float* W2, const float* b2p, const float* noise,
                                const int* col, const int* row,
                                const int* srcp, const int* dstp,
                                float* out, int E) {
    int e = blockIdx.x * blockDim.x + threadIdx.x;
    if (e >= E) return;
    const float* ce = embed + (long long)col[e]  * FD;
    const float* re = embed + (long long)row[e]  * FD;
    const float* se = embed + (long long)srcp[0] * FD;
    const float* de = embed + (long long)dstp[0] * FD;
    float w = b2p[0];
    for (int n = 0; n < HID; ++n) {
        float h = b1[n];
        for (int f = 0; f < FD; ++f) {
            h += ce[f] * W1[f * HID + n] + re[f] * W1[(FD + f) * HID + n]
               + se[f] * W1[(2 * FD + f) * HID + n] + de[f] * W1[(3 * FD + f) * HID + n];
        }
        h = h > 0.f ? h : 0.f;
        w += h * W2[n];
    }
    float u = noise[e];
    float x = logf(u) - log1pf(-u) + w;
    out[e] = 1.f / (1.f + expf(-x));
}

extern "C" void kernel_launch(void* const* d_in, const int* in_sizes, int n_in,
                              void* d_out, int out_size, void* d_ws, size_t ws_size,
                              hipStream_t stream) {
    const float* embed = (const float*)d_in[0];
    const float* W1    = (const float*)d_in[1];
    const float* b1    = (const float*)d_in[2];
    const float* W2    = (const float*)d_in[3];
    const float* b2    = (const float*)d_in[4];
    const float* noise = (const float*)d_in[5];
    const int*   col   = (const int*)d_in[6];
    const int*   row   = (const int*)d_in[7];
    const int*   srcp  = (const int*)d_in[8];
    const int*   dstp  = (const int*)d_in[9];
    float*       out   = (float*)d_out;

    const int E = in_sizes[5];
    const long long nodesF = (long long)in_sizes[0];         // n_nodes * 128

    size_t embH_bytes = (size_t)nodesF * 2;
    size_t off_w1 = (embH_bytes + 255) & ~(size_t)255;
    size_t off_c  = off_w1 + 8192 * sizeof(unsigned int);
    size_t need   = off_c + 256;

    if (ws_size < need) {
        fallback_kernel<<<(E + 255) / 256, 256, 0, stream>>>(
            embed, W1, b1, W2, b2, noise, col, row, srcp, dstp, out, E);
        return;
    }

    _Float16*     embH   = (_Float16*)d_ws;
    unsigned int* w1pack = (unsigned int*)((char*)d_ws + off_w1);
    float*        cvec   = (float*)((char*)d_ws + off_c);

    long long q = nodesF / 4;
    cvt_embed_kernel<<<(int)((q + 255) / 256), 256, 0, stream>>>(embed, embH, nodesF);
    pack_w1_kernel<<<32, 256, 0, stream>>>(W1, w1pack);
    const_vec_kernel<<<1, 64, 0, stream>>>(embed, W1, b1, srcp, dstp, cvec);

    int ntiles = (E + 15) / 16;
    int blocks = (ntiles + WAVES_PER_BLOCK * 4 - 1) / (WAVES_PER_BLOCK * 4); // ~4 tiles/wave
    if (blocks < 1) blocks = 1;
    if (blocks > 8192) blocks = 8192;
    edge_mlp_kernel<<<blocks, 256, 0, stream>>>(
        embH, w1pack, cvec, W2, b2, noise, col, row, out, E, ntiles);
}
